// TrafficEmbedding_69243462746241
// MI455X (gfx1250) — compile-verified, matrix operand order checked by reading, not run on hardware
//
#include <hip/hip_runtime.h>

// ---------------------------------------------------------------------------
// Problem constants
// ---------------------------------------------------------------------------
#define BB   2
#define TT   288
#define NN   1024
#define DD   64
#define SLICES (BB*TT)            // 576
#define OUT_HALF (37748736L)      // 2*288*1024*64

typedef __attribute__((ext_vector_type(16))) __bf16 v16bf;
typedef __attribute__((ext_vector_type(8)))  float  v8f;

union FragA { v16bf v; uint4 q[2]; };
union Pack8 { uint4 q; __bf16 h[8]; };

__device__ __forceinline__ v8f wmma_bf16(const FragA& a, const FragA& b, v8f c) {
    return __builtin_amdgcn_wmma_f32_16x16x32_bf16(false, a.v, false, b.v,
                                                   (short)0, c, false, false);
}

// ---------------------------------------------------------------------------
// K1: degree, dis = (deg+1e-12)^-0.5, power iteration for lambda_max.
//     ws[0] = 2/lambda ; ws[16..16+1024) = dis
// ---------------------------------------------------------------------------
__global__ __launch_bounds__(1024) void prep_graph(const float* __restrict__ A,
                                                   float* __restrict__ wsf) {
    __shared__ float red[1024];
    __shared__ float uvec[1024];
    int tid = threadIdx.x;
    const float* Arow = A + tid * 1024;

    float deg = 0.f;
    for (int j = 0; j < 1024; ++j) deg += Arow[j];
    float dis = rsqrtf(deg + 1e-12f);

    // deterministic pseudo-random init (power iteration converges regardless)
    float v = __sinf((float)(tid + 1) * 12.9898f) * 43758.5453f;
    v = v - floorf(v) - 0.5f;

    red[tid] = v * v; __syncthreads();
    for (int sft = 512; sft > 0; sft >>= 1) {
        if (tid < sft) red[tid] += red[tid + sft];
        __syncthreads();
    }
    float nrm = sqrtf(red[0]);
    v = v / (nrm + 1e-12f);

    float w = 0.f;
    for (int it = 0; it < 20; ++it) {
        __syncthreads();
        uvec[tid] = dis * v;
        __syncthreads();
        float dot = 0.f;
        for (int j = 0; j < 1024; ++j) dot += Arow[j] * uvec[j];
        w = v - dis * dot;                 // L @ v
        red[tid] = w * w; __syncthreads();
        for (int sft = 512; sft > 0; sft >>= 1) {
            if (tid < sft) red[tid] += red[tid + sft];
            __syncthreads();
        }
        nrm = sqrtf(red[0]);
        v = w / (nrm + 1e-12f);
    }
    // lambda = v . (L v)
    __syncthreads();
    uvec[tid] = dis * v;
    __syncthreads();
    float dot = 0.f;
    for (int j = 0; j < 1024; ++j) dot += Arow[j] * uvec[j];
    w = v - dis * dot;
    red[tid] = v * w; __syncthreads();
    for (int sft = 512; sft > 0; sft >>= 1) {
        if (tid < sft) red[tid] += red[tid + sft];
        __syncthreads();
    }
    if (tid == 0) {
        float lam = fminf(fmaxf(red[0], 1e-6f), 2.0f);
        wsf[0] = 2.0f / lam;
    }
    wsf[16 + tid] = dis;
}

// ---------------------------------------------------------------------------
// K2: tildeL = scale*(I - dis_i A dis_j) - I   -> bf16 matrix in workspace
// ---------------------------------------------------------------------------
__global__ __launch_bounds__(256) void build_L(const float* __restrict__ A,
                                               const float* __restrict__ wsf,
                                               __bf16* __restrict__ Lbf) {
    int i = blockIdx.x;
    float scale = wsf[0];
    const float* dis = wsf + 16;
    float di = dis[i];
    for (int j = threadIdx.x; j < 1024; j += 256) {
        float eye = (i == j) ? 1.f : 0.f;
        float l = eye - di * A[i * 1024 + j] * dis[j];
        Lbf[i * 1024 + j] = (__bf16)(scale * l - eye);
    }
}

// ---------------------------------------------------------------------------
// K3: X_te = LN(Xp + E_day + E_week + E_pos)  — one wave per row (memory bound)
// ---------------------------------------------------------------------------
__global__ __launch_bounds__(256) void temporal_ln(
    const float* __restrict__ X, const float* __restrict__ W_in,
    const float* __restrict__ b_in, const float* __restrict__ emb_day,
    const float* __restrict__ emb_week, const float* __restrict__ gamma,
    const float* __restrict__ beta, const int* __restrict__ minute_idx,
    const int* __restrict__ weekday_idx, float* __restrict__ out) {
    int wave = threadIdx.x >> 5, lane = threadIdx.x & 31;
    long row = (long)blockIdx.x * 8 + wave;         // [0, 576*1024)
    int s = (int)(row >> 10);
    int t = s % TT;
    int mi = minute_idx[t] % 288;
    int wi = weekday_idx[t] % 7;

    const float* xr = X + row * 3;
    float x0 = xr[0], x1 = xr[1], x2 = xr[2];

    const float NEG_LOG1E4_D = -9.210340371976184f / 64.0f;
    float vals[2];
#pragma unroll
    for (int h = 0; h < 2; ++h) {
        int col = lane + 32 * h;
        float xp = b_in[col] + x0 * W_in[col] + x1 * W_in[64 + col] +
                   x2 * W_in[128 + col];
        int pair = col >> 1;
        float div = __expf((float)(2 * pair) * NEG_LOG1E4_D);
        float pe = (col & 1) ? __cosf((float)t * div) : __sinf((float)t * div);
        vals[h] = xp + emb_day[mi * 64 + col] + emb_week[wi * 64 + col] + pe;
    }
    float sum = vals[0] + vals[1];
#pragma unroll
    for (int m = 16; m >= 1; m >>= 1) sum += __shfl_xor(sum, m, 32);
    float mean = sum * 0.015625f;
    float d0 = vals[0] - mean, d1 = vals[1] - mean;
    float s2 = d0 * d0 + d1 * d1;
#pragma unroll
    for (int m = 16; m >= 1; m >>= 1) s2 += __shfl_xor(s2, m, 32);
    float inv = rsqrtf(s2 * 0.015625f + 1e-5f);
    out[row * 64 + lane]      = d0 * inv * gamma[lane]      + beta[lane];
    out[row * 64 + lane + 32] = d1 * inv * gamma[lane + 32] + beta[lane + 32];
}

// ---------------------------------------------------------------------------
// K4: per-(b,t) slice:  T1 = L@Xp ; P = Xp@th1 + T1@(2*th2) ;
//     out_s = Xp@(th0-th2) + L@P ; X_sp = LN(Xp + out_s + cheb_bias + E_pos)
//     One 256-thread block (8 waves) per slice. bf16 WMMA 16x16x32.
// ---------------------------------------------------------------------------
__global__ __launch_bounds__(256) void cheb_spatial(
    const float* __restrict__ X, const float* __restrict__ W_in,
    const float* __restrict__ b_in, const float* __restrict__ theta,
    const float* __restrict__ cheb_bias, const float* __restrict__ gamma,
    const float* __restrict__ beta, const __bf16* __restrict__ Lbf,
    float* __restrict__ out) {
    __shared__ __bf16 XpT[64 * 1024];   // Xp transposed [feat][node]  128 KB
    __shared__ __bf16 PT[64 * 1024];    // P  transposed [feat][node]  128 KB
    __shared__ __bf16 th0T[4096];       // (th0 - th2)^T
    __shared__ __bf16 th1T[4096];       // th1^T
    __shared__ __bf16 th2T[4096];       // (2*th2)^T
    __shared__ __bf16 bounce[8 * 1024]; // per-wave 16x64 C->A relayout
    __shared__ float WinS[192], binS[64], cbS[64], gS[64], btS[64];

    int tid = threadIdx.x, wave = tid >> 5, lane = tid & 31;
    int s = blockIdx.x;
    int t = s % TT;

    if (tid < 192) WinS[tid] = W_in[tid];
    if (tid < 64) {
        binS[tid] = b_in[tid]; cbS[tid] = cheb_bias[tid];
        gS[tid] = gamma[tid];  btS[tid] = beta[tid];
    }
    for (int i = tid; i < 4096; i += 256) {
        int ii = i >> 6, oo = i & 63;
        float t0 = theta[i], t1 = theta[4096 + i], t2 = theta[8192 + i];
        th0T[oo * 64 + ii] = (__bf16)(t0 - t2);
        th1T[oo * 64 + ii] = (__bf16)t1;
        th2T[oo * 64 + ii] = (__bf16)(2.0f * t2);
    }
    // stage Xp^T (bf16) into LDS
    for (int m = tid; m < 1024; m += 256) {
        const float* xr = X + ((long)s * 1024 + m) * 3;
        float x0 = xr[0], x1 = xr[1], x2 = xr[2];
        for (int k = 0; k < 64; ++k) {
            float xp = b_in[k] + x0 * W_in[k] + x1 * W_in[64 + k] +
                       x2 * W_in[128 + k];
            XpT[k * 1024 + m] = (__bf16)xp;
        }
    }
    __syncthreads();

    int mRel = lane & 15;
    int kHi  = lane >> 4;
    int rHi  = kHi * 8;
    const v8f vzero = {0.f, 0.f, 0.f, 0.f, 0.f, 0.f, 0.f, 0.f};

    // -------------------- Phase A: T1 tile + P tile ------------------------
    for (int rt = wave; rt < 64; rt += 8) {
        int R0 = rt * 16;
        v8f acc[4] = {vzero, vzero, vzero, vzero};
        for (int k0 = 0; k0 < 1024; k0 += 32) {
            FragA a;
            const __bf16* ap = Lbf + (long)(R0 + mRel) * 1024 + k0 + kHi * 8;
            a.q[0] = *(const uint4*)ap;
            a.q[1] = *(const uint4*)(ap + 16);
#pragma unroll
            for (int c = 0; c < 4; ++c) {
                FragA b;
                const __bf16* bp = &XpT[(c * 16 + mRel) * 1024 + k0 + kHi * 8];
                b.q[0] = *(const uint4*)bp;
                b.q[1] = *(const uint4*)(bp + 16);
                acc[c] = wmma_bf16(a, b, acc[c]);
            }
        }
        // relayout T1 tile (C layout) -> row-major bounce for A-fragment use
#pragma unroll
        for (int c = 0; c < 4; ++c)
#pragma unroll
            for (int r = 0; r < 8; ++r)
                bounce[wave * 1024 + (rHi + r) * 64 + c * 16 + mRel] =
                    (__bf16)acc[c][r];

        // P tile = Xp_tile @ th1 + T1_tile @ (2*th2)
        v8f pac[4] = {vzero, vzero, vzero, vzero};
        const float* xr = X + ((long)s * 1024 + R0 + mRel) * 3;
        float x0 = xr[0], x1 = xr[1], x2 = xr[2];
#pragma unroll
        for (int kk = 0; kk < 64; kk += 32) {
            FragA axp;
#pragma unroll
            for (int e = 0; e < 16; ++e) {
                int k = kk + ((e >> 3) << 4) + (e & 7) + kHi * 8;
                axp.v[e] = (__bf16)(binS[k] + x0 * WinS[k] +
                                    x1 * WinS[64 + k] + x2 * WinS[128 + k]);
            }
            FragA at1;
            const __bf16* t1p = &bounce[wave * 1024 + mRel * 64 + kk + kHi * 8];
            at1.q[0] = *(const uint4*)t1p;
            at1.q[1] = *(const uint4*)(t1p + 16);
#pragma unroll
            for (int c = 0; c < 4; ++c) {
                FragA b1, b2;
                const __bf16* b1p = &th1T[(c * 16 + mRel) * 64 + kk + kHi * 8];
                const __bf16* b2p = &th2T[(c * 16 + mRel) * 64 + kk + kHi * 8];
                b1.q[0] = *(const uint4*)b1p; b1.q[1] = *(const uint4*)(b1p + 16);
                b2.q[0] = *(const uint4*)b2p; b2.q[1] = *(const uint4*)(b2p + 16);
                pac[c] = wmma_bf16(axp, b1, pac[c]);
                pac[c] = wmma_bf16(at1, b2, pac[c]);
            }
        }
        // store P tile transposed: PT[col][node]
#pragma unroll
        for (int c = 0; c < 4; ++c) {
            Pack8 p;
#pragma unroll
            for (int r = 0; r < 8; ++r) p.h[r] = (__bf16)pac[c][r];
            *(uint4*)&PT[(c * 16 + mRel) * 1024 + R0 + rHi] = p.q;
        }
    }
    __syncthreads();

    // per-lane column constants for finalize
    const float NEG_LOG1E4_D = -9.210340371976184f / 64.0f;
    float pe4[4];
#pragma unroll
    for (int c = 0; c < 4; ++c) {
        int col = c * 16 + mRel;
        int pair = col >> 1;
        float div = __expf((float)(2 * pair) * NEG_LOG1E4_D);
        pe4[c] = (col & 1) ? __cosf((float)t * div) : __sinf((float)t * div);
    }

    // -------------------- Phase C: out = L@P + Xp@th0' ; LN ----------------
    for (int rt = wave; rt < 64; rt += 8) {
        int R0 = rt * 16;
        v8f acc[4] = {vzero, vzero, vzero, vzero};
        for (int k0 = 0; k0 < 1024; k0 += 32) {
            FragA a;
            const __bf16* ap = Lbf + (long)(R0 + mRel) * 1024 + k0 + kHi * 8;
            a.q[0] = *(const uint4*)ap;
            a.q[1] = *(const uint4*)(ap + 16);
#pragma unroll
            for (int c = 0; c < 4; ++c) {
                FragA b;
                const __bf16* bp = &PT[(c * 16 + mRel) * 1024 + k0 + kHi * 8];
                b.q[0] = *(const uint4*)bp;
                b.q[1] = *(const uint4*)(bp + 16);
                acc[c] = wmma_bf16(a, b, acc[c]);
            }
        }
        // + Xp_tile @ (th0 - th2)
        const float* xr = X + ((long)s * 1024 + R0 + mRel) * 3;
        float x0 = xr[0], x1 = xr[1], x2 = xr[2];
#pragma unroll
        for (int kk = 0; kk < 64; kk += 32) {
            FragA axp;
#pragma unroll
            for (int e = 0; e < 16; ++e) {
                int k = kk + ((e >> 3) << 4) + (e & 7) + kHi * 8;
                axp.v[e] = (__bf16)(binS[k] + x0 * WinS[k] +
                                    x1 * WinS[64 + k] + x2 * WinS[128 + k]);
            }
#pragma unroll
            for (int c = 0; c < 4; ++c) {
                FragA b0;
                const __bf16* b0p = &th0T[(c * 16 + mRel) * 64 + kk + kHi * 8];
                b0.q[0] = *(const uint4*)b0p; b0.q[1] = *(const uint4*)(b0p + 16);
                acc[c] = wmma_bf16(axp, b0, acc[c]);
            }
        }
        // finalize: residual + bias + PE, LayerNorm per row (half-wave reduce)
#pragma unroll
        for (int r = 0; r < 8; ++r) {
            int row = R0 + rHi + r;
            const float* xr2 = X + ((long)s * 1024 + row) * 3;
            float y0 = xr2[0], y1 = xr2[1], y2 = xr2[2];
            float vals[4];
#pragma unroll
            for (int c = 0; c < 4; ++c) {
                int col = c * 16 + mRel;
                float xp = binS[col] + y0 * WinS[col] + y1 * WinS[64 + col] +
                           y2 * WinS[128 + col];
                vals[c] = acc[c][r] + cbS[col] + pe4[c] + xp;
            }
            float sum = vals[0] + vals[1] + vals[2] + vals[3];
#pragma unroll
            for (int m = 8; m >= 1; m >>= 1) sum += __shfl_xor(sum, m, 32);
            float mean = sum * 0.015625f;
            float s2 = 0.f;
#pragma unroll
            for (int c = 0; c < 4; ++c) {
                float d = vals[c] - mean;
                s2 += d * d;
            }
#pragma unroll
            for (int m = 8; m >= 1; m >>= 1) s2 += __shfl_xor(s2, m, 32);
            float inv = rsqrtf(s2 * 0.015625f + 1e-5f);
            long ob = OUT_HALF + ((long)s * 1024 + row) * 64;
#pragma unroll
            for (int c = 0; c < 4; ++c) {
                int col = c * 16 + mRel;
                out[ob + col] = (vals[c] - mean) * inv * gS[col] + btS[col];
            }
        }
    }
}

// ---------------------------------------------------------------------------
// Host entry
// ---------------------------------------------------------------------------
extern "C" void kernel_launch(void* const* d_in, const int* in_sizes, int n_in,
                              void* d_out, int out_size, void* d_ws,
                              size_t ws_size, hipStream_t stream) {
    (void)in_sizes; (void)n_in; (void)out_size; (void)ws_size;
    const float* X         = (const float*)d_in[0];
    const float* A         = (const float*)d_in[1];
    const float* W_in      = (const float*)d_in[2];
    const float* b_in      = (const float*)d_in[3];
    const float* theta     = (const float*)d_in[4];
    const float* cheb_bias = (const float*)d_in[5];
    const float* emb_day   = (const float*)d_in[6];
    const float* emb_week  = (const float*)d_in[7];
    const float* ln_gamma  = (const float*)d_in[8];
    const float* ln_beta   = (const float*)d_in[9];
    const int*   minute_i  = (const int*)d_in[10];
    const int*   weekday_i = (const int*)d_in[11];
    float* out = (float*)d_out;

    float*  wsf = (float*)d_ws;
    __bf16* Lbf = (__bf16*)((char*)d_ws + 8192);   // 1024*1024 bf16 = 2 MB

    prep_graph<<<1, 1024, 0, stream>>>(A, wsf);
    build_L<<<1024, 256, 0, stream>>>(A, wsf, Lbf);
    temporal_ln<<<SLICES * NN / 8, 256, 0, stream>>>(
        X, W_in, b_in, emb_day, emb_week, ln_gamma, ln_beta, minute_i,
        weekday_i, out);
    cheb_spatial<<<SLICES, 256, 0, stream>>>(X, W_in, b_in, theta, cheb_bias,
                                             ln_gamma, ln_beta, Lbf, out);
}
